// HopfieldSentimentPredictor_58102317580914
// MI455X (gfx1250) — matrix operand order repaired and checked by reading, not compiled, not run on hardware
//
#include <hip/hip_runtime.h>
#include <stdint.h>
#include <stddef.h>

// ---------------------------------------------------------------------------
// HopfieldSentimentPredictor on MI455X (gfx1250, wave32).
// L2-resident problem; bf16 WMMA for input projections + recurrent LSTM GEMM.
// LSTM hidden state in LDS; per-step gate slabs prefetched one step ahead by
// the Tensor Data Mover (tensor_load_to_lds + s_wait_tensorcnt) into a
// double-buffered LDS staging area to keep L2 latency off the critical path.
// ---------------------------------------------------------------------------

#define B_   32
#define S_   512
#define V_   30000
#define E_   300
#define EP   320      // E padded to multiple of 32 (WMMA K)
#define H_   200
#define G4   800      // 4*H
#define ENC_ 400      // 2*H
#define NH_  4
#define DH_  100
#define HP   224      // H padded to multiple of 32 (recurrent K)
#define GP   832      // 4H rows padded to cover N-tile overrun
#define NEGINF   (-1e9f)
#define SCALEF   1000.0f   // 1/0.001

typedef __attribute__((ext_vector_type(16))) __bf16       v16bf;
typedef __attribute__((ext_vector_type(8)))  float        v8f;
typedef __attribute__((ext_vector_type(4)))  unsigned int v4u;
typedef __attribute__((ext_vector_type(8)))  int          v8i;
typedef __attribute__((ext_vector_type(4)))  int          v4i;

union FragBF { v16bf v; int4 q[2]; };

__device__ __forceinline__ __bf16 f2bf(float x) {
  union { float f; uint32_t u; } a; a.f = x;
  uint32_t r = a.u + 0x7FFFu + ((a.u >> 16) & 1u);   // round-nearest-even
  uint16_t h = (uint16_t)(r >> 16);
  __bf16 o; __builtin_memcpy(&o, &h, 2); return o;
}
__device__ __forceinline__ float bf2f(__bf16 x) {
  uint16_t h; __builtin_memcpy(&h, &x, 2);
  union { uint32_t u; float f; } a; a.u = (uint32_t)h << 16;
  return a.f;
}
__device__ __forceinline__ float sigm(float x) { return 1.0f / (1.0f + __expf(-x)); }

// A fragment (16x32 bf16, row-major, leading dim ld). ISA 7.12.2 layout.
__device__ __forceinline__ v16bf load_fragA(const __bf16* base, int ld) {
  int l   = threadIdx.x & 31;
  int row = l & 15;
  int klo = (l & 16) ? 8 : 0;
  FragBF f;
  f.q[0] = *(const int4*)(base + row * ld + klo);
  f.q[1] = *(const int4*)(base + row * ld + klo + 16);
  return f.v;
}

// B fragment (32x16 bf16) where B = W^T, W row-major [N][K], tile at (n0,k0).
__device__ __forceinline__ v16bf load_fragB(const __bf16* w, int ld, int n0, int k0) {
  int l = threadIdx.x & 31;
  int n = n0 + (l & 15);
  int k = k0 + ((l & 16) ? 8 : 0);
  FragBF f;
  f.q[0] = *(const int4*)(w + n * ld + k);
  f.q[1] = *(const int4*)(w + n * ld + k + 16);
  return f.v;
}

__device__ __forceinline__ v8f wmma_bf16(v16bf a, v16bf b, v8f c) {
  return __builtin_amdgcn_wmma_f32_16x16x32_bf16(false, a, false, b, (short)0, c,
                                                 false, false);
}

// TDM: DMA one step's gate slab (2D tile: 32 rows x 800 bf16, row stride
// S_*G4 elements) from global xW into LDS. D# per CDNA5 ISA §8.3/8.4.
// 6-arg builtin form (clang-23 / therock toolchain): the extra int32x8 is the
// unused second descriptor-group pair (2D tensor -> zero-filled).
__device__ __forceinline__ void tdm_issue_xw(const __bf16* gsrc, uint32_t lds_byte) {
  uint64_t ga = (uint64_t)(uintptr_t)gsrc;
  v4u g0;
  g0.x = 1u;                                               // count=1 (valid D#)
  g0.y = lds_byte;                                         // lds_addr
  g0.z = (uint32_t)ga;                                     // global_addr[31:0]
  g0.w = (uint32_t)((ga >> 32) & 0x1FFFFFFu) | (2u << 30); // addr[56:32]|type=2
  v8i g1;
  g1[0] = (int)(1u << 16);                    // wg_mask=0, data_size=1 (2B)
  g1[1] = (int)((G4 & 0xFFFFu) << 16);        // tensor_dim0[15:0] @ bit48
  g1[2] = (int)((G4 >> 16) | ((unsigned)B_ << 16)); // dim0[31:16] | dim1[15:0]
  g1[3] = (int)((unsigned)G4 << 16);          // dim1[31:16]=0 | tile_dim0=800
  g1[4] = (int)B_;                            // tile_dim1=32, tile_dim2=0
  g1[5] = (int)((unsigned)(S_ * G4));         // tensor_dim0_stride[31:0]
  g1[6] = 0;                                  // stride[47:32] | dim1_stride lo
  g1[7] = 0;
  v4i z4 = {0, 0, 0, 0};
  v8i z8 = {0, 0, 0, 0, 0, 0, 0, 0};
  __builtin_amdgcn_tensor_load_to_lds(g0, g1, z4, z4, z8, 0);
}

// ---------------------------------------------------------------------------
// Small prep kernels
// ---------------------------------------------------------------------------

__global__ void zero_f32(float* p, size_t n) {
  size_t i = (size_t)blockIdx.x * blockDim.x + threadIdx.x;
  if (i < n) p[i] = 0.0f;
}

__global__ __launch_bounds__(256) void lengths_kernel(const unsigned char* mask,
                                                      int* lengths) {
  __shared__ int red[256];
  int b = blockIdx.x;
  int acc = 0;
  for (int s = threadIdx.x; s < S_; s += blockDim.x)
    acc += mask[b * S_ + s] ? 1 : 0;
  red[threadIdx.x] = acc;
  __syncthreads();
  for (int st = 128; st > 0; st >>= 1) {
    if ((int)threadIdx.x < st) red[threadIdx.x] += red[threadIdx.x + st];
    __syncthreads();
  }
  if (threadIdx.x == 0) lengths[b] = red[0];
}

__global__ void convert_pad(const float* src, __bf16* dst, int rows, int scols,
                            int drows, int dcols) {
  int i = blockIdx.x * blockDim.x + threadIdx.x;
  if (i >= drows * dcols) return;
  int r = i / dcols, c = i % dcols;
  float v = (r < rows && c < scols) ? src[r * scols + c] : 0.0f;
  dst[i] = f2bf(v);
}

// q = q_state @ W_q ; kq[h] = W_k[:, h*DH:+DH] @ q[h] ; wvec = W_out @ W_head
__global__ __launch_bounds__(512) void prep_small(
    const float* __restrict__ qs, const float* __restrict__ Wq,
    const float* __restrict__ Wk, const float* __restrict__ Wout,
    const float* __restrict__ Whead, float* q, float* kq, float* wvec) {
  __shared__ float ql[ENC_];
  int t = threadIdx.x;
  if (t < ENC_) {
    float acc = 0.0f;
    for (int i = 0; i < ENC_; ++i) acc += qs[i] * Wq[i * ENC_ + t];
    ql[t] = acc;
    q[t]  = acc;
    float wa = 0.0f;
    for (int j = 0; j < ENC_; ++j) wa += Wout[t * ENC_ + j] * Whead[j];
    wvec[t] = wa;
  }
  __syncthreads();
  for (int idx = t; idx < NH_ * ENC_; idx += blockDim.x) {
    int h = idx / ENC_, e = idx % ENC_;
    float acc = 0.0f;
    for (int d = 0; d < DH_; ++d) acc += Wk[e * ENC_ + h * DH_ + d] * ql[h * DH_ + d];
    kq[idx] = acc;
  }
}

// Gather embeddings (forward + per-sample-reversed), bf16, pad E 300->320.
__global__ __launch_bounds__(320) void gather_embed(
    const int* __restrict__ x, const int* __restrict__ lengths,
    const float* __restrict__ table, __bf16* __restrict__ embf,
    __bf16* __restrict__ embb) {
  int bs = blockIdx.x;
  int b = bs / S_, t = bs % S_;
  int c = threadIdx.x;
  int len = lengths[b];
  int src = len - 1 - t;
  src = src < 0 ? 0 : (src > S_ - 1 ? S_ - 1 : src);
  __bf16 vf = f2bf(0.0f), vb = f2bf(0.0f);
  if (c < E_) {
    vf = f2bf(table[(size_t)x[b * S_ + t] * E_ + c]);
    vb = f2bf(table[(size_t)x[b * S_ + src] * E_ + c]);
  }
  embf[(size_t)bs * EP + c] = vf;
  embb[(size_t)bs * EP + c] = vb;
}

// ---------------------------------------------------------------------------
// Input projection GEMM: out[16384][800](bf16) = A[16384][320] @ W^T + bias.
// 2x2 tile block per wave (32x32 output): 4 WMMAs per 4 fragment loads.
// ---------------------------------------------------------------------------
__global__ __launch_bounds__(256) void gemm_xw(
    const __bf16* __restrict__ A, const __bf16* __restrict__ W,
    const float* __restrict__ bias, __bf16* __restrict__ out) {
  const int M = B_ * S_;
  const int ng = G4 / 32;                   // 25 col groups
  const int total = (M / 32) * ng;          // 12800 wave jobs
  int wid = blockIdx.x * (blockDim.x >> 5) + (threadIdx.x >> 5);
  if (wid >= total) return;
  int mrow = (wid / ng) * 32;
  int ncol = (wid % ng) * 32;
  int l  = threadIdx.x & 31;
  int c0 = ncol + (l & 15), c1 = c0 + 16;
  int hi = (l >> 4) & 1;

  v8f a00, a01, a10, a11;
  float b0v = bias[c0], b1v = bias[c1];
#pragma unroll
  for (int v = 0; v < 8; ++v) { a00[v] = b0v; a01[v] = b1v; a10[v] = b0v; a11[v] = b1v; }

  const __bf16* A0 = A + (size_t)mrow * EP;
  const __bf16* A1 = A0 + 16 * EP;
#pragma unroll 2
  for (int kk = 0; kk < EP; kk += 32) {
    if (kk + 32 < EP) {   // next K-block prefetch (global_prefetch_b8)
      __builtin_prefetch(A0 + kk + 32, 0, 1);
      __builtin_prefetch(A1 + kk + 32, 0, 1);
      __builtin_prefetch(W + (size_t)ncol * EP + kk + 32, 0, 1);
    }
    v16bf fa0 = load_fragA(A0 + kk, EP);
    v16bf fa1 = load_fragA(A1 + kk, EP);
    v16bf fb0 = load_fragB(W, EP, ncol, kk);
    v16bf fb1 = load_fragB(W, EP, ncol + 16, kk);
    a00 = wmma_bf16(fa0, fb0, a00);
    a01 = wmma_bf16(fa0, fb1, a01);
    a10 = wmma_bf16(fa1, fb0, a10);
    a11 = wmma_bf16(fa1, fb1, a11);
  }
#pragma unroll
  for (int v = 0; v < 8; ++v) {
    int r0 = mrow + v + hi * 8;
    int r1 = r0 + 16;
    out[(size_t)r0 * G4 + c0] = f2bf(a00[v]);
    out[(size_t)r0 * G4 + c1] = f2bf(a01[v]);
    out[(size_t)r1 * G4 + c0] = f2bf(a10[v]);
    out[(size_t)r1 * G4 + c1] = f2bf(a11[v]);
  }
}

// ---------------------------------------------------------------------------
// Persistent LSTM scan: one WG per direction, 26 waves (2 batch x 13 col
// tiles). h in LDS (bf16), c in VGPRs, gate slabs TDM-prefetched into a
// double-buffered LDS staging area one step ahead.
// ---------------------------------------------------------------------------
__global__ __launch_bounds__(832) void lstm_scan(
    const __bf16* __restrict__ xWf, const __bf16* __restrict__ xWb,
    const __bf16* __restrict__ whhf, const __bf16* __restrict__ whhb,
    const int* __restrict__ lengths, float* __restrict__ hout) {
  const int backward = blockIdx.x;
  const __bf16* xW  = backward ? xWb : xWf;
  const __bf16* whh = backward ? whhb : whhf;

  __shared__ __align__(16) __bf16 hbuf[B_ * HP];        // 14336 B
  __shared__ __align__(16) __bf16 xbuf[2][B_ * G4];     // 2 x 51200 B staging
  for (int i = threadIdx.x; i < B_ * HP; i += blockDim.x) hbuf[i] = f2bf(0.0f);

  int wv = threadIdx.x >> 5;       // 0..25
  int mt = wv / 13;                // batch tile
  int nt = wv % 13;                // h-column tile (covers 208 cols)
  int l  = threadIdx.x & 31;
  int col = nt * 16 + (l & 15);
  int hi  = (l >> 4) & 1;
  bool colok = col < H_;
  int cc = colok ? col : 0;

  uint32_t xlds0 = (uint32_t)(uintptr_t)(&xbuf[0][0]);
  uint32_t xlds1 = (uint32_t)(uintptr_t)(&xbuf[1][0]);

  // Stage step 0 gates via TDM (wave 0 only; EXEC ignored by tensor ops).
  if (wv == 0) {
    tdm_issue_xw(xW, xlds0);
    __builtin_amdgcn_s_wait_tensorcnt(0);
  }
  __syncthreads();   // hbuf zeroed + step-0 gates staged

  int rowv[8], lenv[8];
#pragma unroll
  for (int v = 0; v < 8; ++v) {
    rowv[v] = mt * 16 + v + hi * 8;
    lenv[v] = lengths[rowv[v]];
  }

  v8f c;
  float hprev[8];
#pragma unroll
  for (int v = 0; v < 8; ++v) { c[v] = 0.0f; hprev[v] = 0.0f; }

  const __bf16* Abase = hbuf + mt * 16 * HP;

  for (int t = 0; t < S_; ++t) {
    const int cb = t & 1;
    // gate accumulators from LDS-staged x-projection
    v8f ai, af, ag, ao;
#pragma unroll
    for (int v = 0; v < 8; ++v) {
      const __bf16* xr = &xbuf[cb][rowv[v] * G4];
      ai[v] = bf2f(xr[0 * H_ + cc]);
      af[v] = bf2f(xr[1 * H_ + cc]);
      ag[v] = bf2f(xr[2 * H_ + cc]);
      ao[v] = bf2f(xr[3 * H_ + cc]);
    }
    // kick off DMA of next step's gates while WMMAs run
    if (wv == 0 && t + 1 < S_)
      tdm_issue_xw(xW + (size_t)(t + 1) * G4, cb ? xlds0 : xlds1);

    // gates += h_prev @ W_hh^T   (A fragments from LDS)
#pragma unroll
    for (int kk = 0; kk < HP; kk += 32) {
      v16bf a  = load_fragA(Abase + kk, HP);
      v16bf b0 = load_fragB(whh, HP, 0 * H_ + nt * 16, kk);
      ai = wmma_bf16(a, b0, ai);
      v16bf b1 = load_fragB(whh, HP, 1 * H_ + nt * 16, kk);
      af = wmma_bf16(a, b1, af);
      v16bf b2 = load_fragB(whh, HP, 2 * H_ + nt * 16, kk);
      ag = wmma_bf16(a, b2, ag);
      v16bf b3 = load_fragB(whh, HP, 3 * H_ + nt * 16, kk);
      ao = wmma_bf16(a, b3, ao);
    }
    // pointwise LSTM + masked state update + output scatter
    float hnew[8];
#pragma unroll
    for (int v = 0; v < 8; ++v) {
      float cn = sigm(af[v]) * c[v] + sigm(ai[v]) * tanhf(ag[v]);
      float hn = sigm(ao[v]) * tanhf(cn);
      bool m = t < lenv[v];
      c[v]    = m ? cn : c[v];
      hnew[v] = m ? hn : hprev[v];
      if (m && colok) {
        int pos = backward ? (lenv[v] - 1 - t) : t;
        hout[((size_t)rowv[v] * S_ + pos) * ENC_ + (backward ? H_ : 0) + col] = hn;
      }
    }
    __syncthreads();   // all waves done reading hbuf/xbuf for this step
    if (colok) {
#pragma unroll
      for (int v = 0; v < 8; ++v) {
        hbuf[rowv[v] * HP + col] = f2bf(hnew[v]);
        hprev[v] = hnew[v];
      }
    }
    if (wv == 0 && t + 1 < S_)
      __builtin_amdgcn_s_wait_tensorcnt(0);   // next step's gates landed
    __syncthreads();   // publish hbuf + staged gates to all waves
  }
}

// ---------------------------------------------------------------------------
// scores[b,h,s] = SCALING * h[b,s,:]·kq[h,:]   (masked)
// ---------------------------------------------------------------------------
__global__ __launch_bounds__(512) void scores_kernel(
    const float* __restrict__ hout, const float* __restrict__ kq,
    const int* __restrict__ lengths, float* __restrict__ scores) {
  int bh = blockIdx.x;
  int b = bh / NH_, h = bh % NH_;
  int s = threadIdx.x;
  const float* hr = hout + ((size_t)b * S_ + s) * ENC_;
  const float* kr = kq + h * ENC_;
  float acc = 0.0f;
#pragma unroll 4
  for (int e = 0; e < ENC_; ++e) acc += hr[e] * kr[e];
  scores[bh * S_ + s] = (s < lengths[b]) ? acc * SCALEF : NEGINF;
}

// ---------------------------------------------------------------------------
// SparseMAP capped-simplex projection, one wave per (b,head).
// ---------------------------------------------------------------------------
__global__ __launch_bounds__(32) void sparsemap_kernel(
    const float* __restrict__ scores, const int* __restrict__ lengths,
    float* __restrict__ p) {
  int bh = blockIdx.x;
  int b = bh / NH_;
  int l = threadIdx.x;
  float th[16];
#pragma unroll
  for (int j = 0; j < 16; ++j) th[j] = scores[bh * S_ + j * 32 + l];
  float k = rintf(0.2f * (float)lengths[b]);

  float mn = th[0], mx = th[0];
#pragma unroll
  for (int j = 1; j < 16; ++j) { mn = fminf(mn, th[j]); mx = fmaxf(mx, th[j]); }
  for (int m = 16; m > 0; m >>= 1) {
    mn = fminf(mn, __shfl_xor(mn, m, 32));
    mx = fmaxf(mx, __shfl_xor(mx, m, 32));
  }
  float lo = mn - 1.0f, hi = mx;
  for (int it = 0; it < 60; ++it) {
    float mid = 0.5f * (lo + hi);
    float s = 0.0f;
#pragma unroll
    for (int j = 0; j < 16; ++j) s += fminf(fmaxf(th[j] - mid, 0.0f), 1.0f);
    for (int m = 16; m > 0; m >>= 1) s += __shfl_xor(s, m, 32);
    bool big = s > k;
    lo = big ? mid : lo;
    hi = big ? hi : mid;
  }
  float tau0 = 0.5f * (lo + hi);
  float nU = 0.0f, nS = 0.0f, sf = 0.0f;
#pragma unroll
  for (int j = 0; j < 16; ++j) {
    float mu = th[j] - tau0;
    if (mu >= 1.0f)      nU += 1.0f;
    else if (mu > 0.0f) { nS += 1.0f; sf += th[j]; }
  }
  for (int m = 16; m > 0; m >>= 1) {
    nU += __shfl_xor(nU, m, 32);
    nS += __shfl_xor(nS, m, 32);
    sf += __shfl_xor(sf, m, 32);
  }
  float tau = (sf + nU - k) / fmaxf(nS, 1.0f);
  tau = (nS > 0.0f) ? tau : tau0;
#pragma unroll
  for (int j = 0; j < 16; ++j)
    p[bh * S_ + j * 32 + l] = fminf(fmaxf(th[j] - tau, 0.0f), 1.0f);
}

__global__ __launch_bounds__(512) void pool_kernel(
    const float* __restrict__ p, const float* __restrict__ hout,
    float* __restrict__ wsum) {
  int bh = blockIdx.x;
  int b = bh / NH_;
  int e = threadIdx.x;
  if (e >= ENC_) return;
  float acc = 0.0f;
  for (int s = 0; s < S_; ++s)
    acc += p[bh * S_ + s] * hout[((size_t)b * S_ + s) * ENC_ + e];
  wsum[bh * ENC_ + e] = acc;
}

__global__ __launch_bounds__(512) void pooledpre_kernel(
    const float* __restrict__ wsum, const float* __restrict__ Wv,
    float* __restrict__ pooled) {
  int b = blockIdx.x;
  int j = threadIdx.x;
  if (j >= ENC_) return;
  int h = j / DH_;
  const float* w = wsum + (b * NH_ + h) * ENC_;
  float acc = 0.0f;
  for (int e = 0; e < ENC_; ++e) acc += w[e] * Wv[e * ENC_ + j];
  pooled[b * ENC_ + j] = acc;
}

__global__ __launch_bounds__(512) void y_kernel(
    const float* __restrict__ pooled, const float* __restrict__ wvec,
    const float* __restrict__ bhead, float* __restrict__ y) {
  __shared__ float red[512];
  int b = blockIdx.x, j = threadIdx.x;
  red[j] = (j < ENC_) ? pooled[b * ENC_ + j] * wvec[j] : 0.0f;
  __syncthreads();
  for (int st = 256; st > 0; st >>= 1) {
    if (j < st) red[j] += red[j + st];
    __syncthreads();
  }
  if (j == 0) y[b] = sigm(red[0] + bhead[0]);
}

__global__ void z_kernel(const float* __restrict__ p,
                         const int* __restrict__ lengths, float* __restrict__ z) {
  int i = blockIdx.x * blockDim.x + threadIdx.x;
  if (i >= B_ * S_) return;
  int b = i / S_, s = i % S_;
  float v = 0.0f;
  if (s < lengths[b])
    v = 0.25f * (p[(b * NH_ + 0) * S_ + s] + p[(b * NH_ + 1) * S_ + s] +
                 p[(b * NH_ + 2) * S_ + s] + p[(b * NH_ + 3) * S_ + s]);
  z[i] = v;
}

// ---------------------------------------------------------------------------
// Launch
// ---------------------------------------------------------------------------
extern "C" void kernel_launch(void* const* d_in, const int* in_sizes, int n_in,
                              void* d_out, int out_size, void* d_ws,
                              size_t ws_size, hipStream_t stream) {
  (void)in_sizes; (void)n_in; (void)out_size; (void)ws_size;
  const int*           x      = (const int*)d_in[0];
  const unsigned char* mask   = (const unsigned char*)d_in[1];
  const float*         table  = (const float*)d_in[2];
  const float*         wihf   = (const float*)d_in[3];
  const float*         whhf   = (const float*)d_in[4];
  const float*         bf     = (const float*)d_in[5];
  const float*         wihb   = (const float*)d_in[6];
  const float*         whhb   = (const float*)d_in[7];
  const float*         bb     = (const float*)d_in[8];
  const float*         qstate = (const float*)d_in[9];
  const float*         Wq     = (const float*)d_in[10];
  const float*         Wk     = (const float*)d_in[11];
  const float*         Wv     = (const float*)d_in[12];
  const float*         Wout   = (const float*)d_in[13];
  const float*         Whead  = (const float*)d_in[14];
  const float*         bhead  = (const float*)d_in[15];

  float* y = (float*)d_out;        // [32]
  float* z = y + B_;               // [32*512]

  uint8_t* base = (uint8_t*)d_ws;
  size_t cur = 0;
  auto alloc = [&](size_t bytes) -> void* {
    cur = (cur + 255) & ~(size_t)255;
    void* p = base + cur;
    cur += bytes;
    return p;
  };
  int*    lengths = (int*)alloc(B_ * sizeof(int));
  __bf16* embf    = (__bf16*)alloc((size_t)B_ * S_ * EP * 2);
  __bf16* embb    = (__bf16*)alloc((size_t)B_ * S_ * EP * 2);
  __bf16* wihf_b  = (__bf16*)alloc((size_t)G4 * EP * 2);
  __bf16* wihb_b  = (__bf16*)alloc((size_t)G4 * EP * 2);
  __bf16* whhf_b  = (__bf16*)alloc((size_t)GP * HP * 2);
  __bf16* whhb_b  = (__bf16*)alloc((size_t)GP * HP * 2);
  __bf16* xWf     = (__bf16*)alloc((size_t)B_ * S_ * G4 * 2);
  __bf16* xWb     = (__bf16*)alloc((size_t)B_ * S_ * G4 * 2);
  float*  hout    = (float*)alloc((size_t)B_ * S_ * ENC_ * 4);
  float*  qv      = (float*)alloc(ENC_ * 4);
  float*  kq      = (float*)alloc(NH_ * ENC_ * 4);
  float*  wvec    = (float*)alloc(ENC_ * 4);
  float*  scores  = (float*)alloc((size_t)B_ * NH_ * S_ * 4);
  float*  pbuf    = (float*)alloc((size_t)B_ * NH_ * S_ * 4);
  float*  wsum    = (float*)alloc((size_t)B_ * NH_ * ENC_ * 4);
  float*  pooled  = (float*)alloc((size_t)B_ * ENC_ * 4);

  {
    size_t n = (size_t)B_ * S_ * ENC_;
    zero_f32<<<(unsigned)((n + 255) / 256), 256, 0, stream>>>(hout, n);
  }
  lengths_kernel<<<B_, 256, 0, stream>>>(mask, lengths);
  {
    int n;
    n = G4 * EP;
    convert_pad<<<(n + 255) / 256, 256, 0, stream>>>(wihf, wihf_b, G4, E_, G4, EP);
    convert_pad<<<(n + 255) / 256, 256, 0, stream>>>(wihb, wihb_b, G4, E_, G4, EP);
    n = GP * HP;
    convert_pad<<<(n + 255) / 256, 256, 0, stream>>>(whhf, whhf_b, G4, H_, GP, HP);
    convert_pad<<<(n + 255) / 256, 256, 0, stream>>>(whhb, whhb_b, G4, H_, GP, HP);
  }
  prep_small<<<1, 512, 0, stream>>>(qstate, Wq, Wk, Wout, Whead, qv, kq, wvec);
  gather_embed<<<B_ * S_, 320, 0, stream>>>(x, lengths, table, embf, embb);

  {
    const int tiles = (B_ * S_ / 32) * (G4 / 32);    // 12800 wave jobs
    const int blocks = (tiles + 7) / 8;              // 8 waves/block
    gemm_xw<<<blocks, 256, 0, stream>>>(embf, wihf_b, bf, xWf);
    gemm_xw<<<blocks, 256, 0, stream>>>(embb, wihb_b, bb, xWb);
  }

  lstm_scan<<<2, 832, 0, stream>>>(xWf, xWb, whhf_b, whhb_b, lengths, hout);

  scores_kernel<<<B_ * NH_, 512, 0, stream>>>(hout, kq, lengths, scores);
  sparsemap_kernel<<<B_ * NH_, 32, 0, stream>>>(scores, lengths, pbuf);
  pool_kernel<<<B_ * NH_, 512, 0, stream>>>(pbuf, hout, wsum);
  pooledpre_kernel<<<B_, 512, 0, stream>>>(wsum, Wv, pooled);
  y_kernel<<<B_, 512, 0, stream>>>(pooled, wvec, bhead, y);
  z_kernel<<<(B_ * S_ + 255) / 256, 256, 0, stream>>>(pbuf, lengths, z);
}
